// GraphTransformerLayer_grok_40132174414147
// MI455X (gfx1250) — compile-verified, matrix operand order checked by reading, not run on hardware
//
#include <hip/hip_runtime.h>

typedef __bf16 bf16_t;
typedef __attribute__((ext_vector_type(16))) __bf16 v16bf;
typedef __attribute__((ext_vector_type(8)))  __bf16 v8bf;
typedef __attribute__((ext_vector_type(8)))  float  v8f;
typedef int v4i_gcc __attribute__((vector_size(4 * sizeof(int))));

#define N_NODES 4096
#define DMODEL  256
#define NHEAD   4
#define DHEAD   64
#define DEG     32
#define DFF     1024

#define KSTEP 64
#define LROW  72   // LDS row pitch in bf16 elems (64 + 8 pad -> 36-bank stride, 16B aligned)

// CDNA5 async global->LDS staging (ASYNCcnt path); guarded so either toolchain compiles.
// Probe result (round 2 diagnostic): builtin exists with 4 args and takes
// generic pointers to v4i (vector_size(16) int), not addrspace-qualified void*.
#if defined(__AMDGCN__) && __has_builtin(__builtin_amdgcn_global_load_async_to_lds_b128) && \
    __has_builtin(__builtin_amdgcn_s_wait_asynccnt)
#define USE_ASYNC_LDS 1
#define COPY16(gsrc, ldst)                                                        \
  __builtin_amdgcn_global_load_async_to_lds_b128(                                 \
      (v4i_gcc*)(gsrc), (v4i_gcc*)(ldst), 0, 0)
#define WAIT_ASYNC(n) __builtin_amdgcn_s_wait_asynccnt(n)
#else
#define USE_ASYNC_LDS 0
#define COPY16(gsrc, ldst) (*(v8bf*)(ldst) = *(const v8bf*)(gsrc))
#define WAIT_ASYNC(n) ((void)0)
#endif

// ---------------------------------------------------------------------------
// fp32 -> bf16 elementwise conversion
// ---------------------------------------------------------------------------
__global__ void f32_to_bf16_kernel(const float* __restrict__ x, bf16_t* __restrict__ y, int n) {
  int i = blockIdx.x * blockDim.x + threadIdx.x;
  if (i < n) y[i] = (bf16_t)x[i];
}

__global__ void pack_bias3_kernel(const float* __restrict__ a, const float* __restrict__ b,
                                  const float* __restrict__ c, float* __restrict__ o) {
  int i = blockIdx.x * blockDim.x + threadIdx.x;
  if (i < 3 * DMODEL) o[i] = (i < DMODEL) ? a[i] : ((i < 2 * DMODEL) ? b[i - DMODEL] : c[i - 2 * DMODEL]);
}

// ---------------------------------------------------------------------------
// WMMA GEMM:  out[M, Ncols] = A[M, K](bf16) * W[Ncols, K]^T (bf16) + epilogue
//
// Block = 256 threads (8 waves). Block tile = 64 rows x 128 cols.
// Double-buffered LDS pipeline, K-step 64:
//   A tile 64x64 bf16 and W tile 128x64 bf16 staged via async global->LDS
//   (ASYNCcnt) while the previous tile is consumed by WMMA from LDS.
// Wave tile = 32x32 (2x2 of v_wmma_f32_16x16x32_bf16); 8 wmma per K-step.
// MODE bits: 1=relu, 2=+resid (width Ncols), 4=store fp32, 8=store bf16
// ---------------------------------------------------------------------------
template <int MODE>
__launch_bounds__(256)
__global__ void gemm_bf16_wmma(const bf16_t* __restrict__ A, const bf16_t* __restrict__ W,
                               const float* __restrict__ bias, const float* __restrict__ resid,
                               float* __restrict__ outF, bf16_t* __restrict__ outB,
                               int Ncols, int K) {
  __shared__ bf16_t lsA[2][64 * LROW];
  __shared__ bf16_t lsW[2][128 * LROW];

  const int tid  = threadIdx.x;
  const int wave = tid >> 5;
  const int lane = tid & 31;
  const int l    = lane & 15;   // row (A) / col (B) within 16
  const int hi   = lane >> 4;   // K-half selector per ISA 16-bit layout
  const int rB   = blockIdx.y * 64;    // block row base
  const int cB   = blockIdx.x * 128;   // block col base
  const int r0w  = (wave & 1) * 32;    // wave offsets inside the block tile
  const int c0w  = (wave >> 1) * 32;

  const v8f zero = {0.f, 0.f, 0.f, 0.f, 0.f, 0.f, 0.f, 0.f};
  v8f acc[2][2] = {{zero, zero}, {zero, zero}};

  // Staging geometry: each thread moves 16B chunks (8 bf16).
  const int sRow = tid >> 3;        // 0..31, + r*32
  const int sKc  = (tid & 7) * 8;   // k-chunk within the 64-wide K slab

  auto stage = [&](int s, int buf) {
    const int k0 = s * KSTEP;
#pragma unroll
    for (int r = 0; r < 2; ++r) {   // 64 A rows
      const int row = sRow + r * 32;
      COPY16(A + (size_t)(rB + row) * K + k0 + sKc, &lsA[buf][row * LROW + sKc]);
    }
#pragma unroll
    for (int r = 0; r < 4; ++r) {   // 128 W rows
      const int row = sRow + r * 32;
      COPY16(W + (size_t)(cB + row) * K + k0 + sKc, &lsW[buf][row * LROW + sKc]);
    }
  };

  const int steps = K / KSTEP;
  stage(0, 0);

  for (int s = 0; s < steps; ++s) {
    const int cur = s & 1;
    if (s + 1 < steps) {
      stage(s + 1, cur ^ 1);
      WAIT_ASYNC(6);   // 6 newer copies outstanding -> this step's tile landed
    } else {
      WAIT_ASYNC(0);
    }
    __syncthreads();

    const bf16_t* tA = lsA[cur];
    const bf16_t* tW = lsW[cur];
#pragma unroll
    for (int ks = 0; ks < KSTEP; ks += 32) {
      union V16u { v16bf v; v8bf h[2]; };
      v16bf afrag[2], bfrag[2];
#pragma unroll
      for (int mi = 0; mi < 2; ++mi) {
        // 16-bit A layout: lanes 0-15 hold K 0..7 / 16..23, lanes 16-31 hold 8..15 / 24..31
        const bf16_t* p = tA + (r0w + mi * 16 + l) * LROW + ks + hi * 8;
        V16u t;
        t.h[0] = *(const v8bf*)p;
        t.h[1] = *(const v8bf*)(p + 16);
        afrag[mi] = t.v;
      }
#pragma unroll
      for (int ni = 0; ni < 2; ++ni) {
        // B[k, n] = W[n, k]  (out = A * W^T): same K pattern on W rows
        const bf16_t* p = tW + (c0w + ni * 16 + l) * LROW + ks + hi * 8;
        V16u t;
        t.h[0] = *(const v8bf*)p;
        t.h[1] = *(const v8bf*)(p + 16);
        bfrag[ni] = t.v;
      }
#pragma unroll
      for (int mi = 0; mi < 2; ++mi)
#pragma unroll
        for (int ni = 0; ni < 2; ++ni)
          acc[mi][ni] = __builtin_amdgcn_wmma_f32_16x16x32_bf16(
              false, afrag[mi], false, bfrag[ni], (short)0, acc[mi][ni], false, false);
    }
    __syncthreads();
  }

  // Epilogue: C/D layout: VGPR j -> row +hi*8+j, lane l -> col +l
#pragma unroll
  for (int mi = 0; mi < 2; ++mi) {
#pragma unroll
    for (int ni = 0; ni < 2; ++ni) {
      const int col  = cB + c0w + ni * 16 + l;
      const float bv = bias[col];
#pragma unroll
      for (int j = 0; j < 8; ++j) {
        const int row = rB + r0w + mi * 16 + hi * 8 + j;
        float y = acc[mi][ni][j] + bv;
        if (MODE & 2) y += resid[(size_t)row * Ncols + col];
        if (MODE & 1) y = fmaxf(y, 0.f);
        if (MODE & 4) outF[(size_t)row * Ncols + col] = y;
        if (MODE & 8) outB[(size_t)row * Ncols + col] = (bf16_t)y;
      }
    }
  }
}

// ---------------------------------------------------------------------------
// Sparse graph attention: one wave per (node, head); lane = one outgoing edge.
// qkv layout: [N, 768] bf16, cols 0-255 = q, 256-511 = k, 512-767 = v.
// ---------------------------------------------------------------------------
__launch_bounds__(128)
__global__ void attn_sparse_kernel(const bf16_t* __restrict__ qkv,
                                   const float* __restrict__ ew,
                                   const int* __restrict__ dst,
                                   bf16_t* __restrict__ attn) {
  const int wv   = blockIdx.x * 4 + (threadIdx.x >> 5);
  const int lane = threadIdx.x & 31;
  const int n    = wv >> 2;
  const int head = wv & 3;

  const int   e = n * DEG + lane;     // src = repeat(arange(N), DEG)
  const int   d = dst[e];
  const float w = fmaxf(ew[e], 0.f);

  // score = q[n,head,:] . k[d,head,:] / sqrt(64)
  const v8bf* qp = (const v8bf*)(qkv + (size_t)n * 768 + head * DHEAD);
  const v8bf* kp = (const v8bf*)(qkv + (size_t)d * 768 + 256 + head * DHEAD);
  float s = 0.f;
#pragma unroll
  for (int c = 0; c < 8; ++c) {
    v8bf qv = qp[c], kv = kp[c];
#pragma unroll
    for (int i = 0; i < 8; ++i) s += (float)qv[i] * (float)kv[i];
  }
  float logit = s * 0.125f + w + 1e-8f;

  // Dense mask .at[src,dst].set semantics: duplicate dst collapses to one
  // entry (keep last occurrence).
  bool dead = false;
  for (int j = 0; j < 32; ++j) {
    int dj = __shfl(d, j, 32);
    if (dj == d && j > lane) dead = true;
  }
  if (dead) logit = -3.0e38f;

  // wave-wide softmax over the 32 edges
  float m = logit;
  for (int off = 16; off > 0; off >>= 1) m = fmaxf(m, __shfl_xor(m, off, 32));
  float p = __expf(logit - m);
  float sum = p;
  for (int off = 16; off > 0; off >>= 1) sum += __shfl_xor(sum, off, 32);
  p /= sum;

  // attn_out[n, head, :] = sum_j p_j * v[dst_j, head, :]; lane owns dims {lane, lane+32}
  float ax = 0.f, ay = 0.f;
  for (int j = 0; j < 32; ++j) {
    float pj = __shfl(p, j, 32);
    int   dj = __shfl(d, j, 32);
    const bf16_t* vp = qkv + (size_t)dj * 768 + 512 + head * DHEAD;
    ax += pj * (float)vp[lane];
    ay += pj * (float)vp[lane + 32];
  }
  bf16_t* op = attn + (size_t)n * DMODEL + head * DHEAD;
  op[lane]      = (bf16_t)ax;
  op[lane + 32] = (bf16_t)ay;
}

// ---------------------------------------------------------------------------
// LayerNorm over rows of width 256; one wave per row (8 elems/lane).
// ---------------------------------------------------------------------------
__launch_bounds__(256)
__global__ void layernorm_kernel(const float* __restrict__ x,
                                 const float* __restrict__ g,
                                 const float* __restrict__ b,
                                 float* __restrict__ outF,
                                 bf16_t* __restrict__ outB) {
  const int row  = blockIdx.x * 8 + (threadIdx.x >> 5);
  const int lane = threadIdx.x & 31;
  const float* xr = x + (size_t)row * DMODEL + lane * 8;
  float v[8];
  float s = 0.f, ss = 0.f;
#pragma unroll
  for (int i = 0; i < 8; ++i) { v[i] = xr[i]; s += v[i]; ss += v[i] * v[i]; }
  for (int off = 16; off > 0; off >>= 1) {
    s  += __shfl_xor(s, off, 32);
    ss += __shfl_xor(ss, off, 32);
  }
  const float mu  = s * (1.f / DMODEL);
  const float var = ss * (1.f / DMODEL) - mu * mu;
  const float rs  = rsqrtf(var + 1e-5f);
#pragma unroll
  for (int i = 0; i < 8; ++i) {
    const int c   = lane * 8 + i;
    const float y = (v[i] - mu) * rs * g[c] + b[c];
    if (outF) outF[(size_t)row * DMODEL + c] = y;
    if (outB) outB[(size_t)row * DMODEL + c] = (bf16_t)y;
  }
}

// ---------------------------------------------------------------------------
// Launch
// ---------------------------------------------------------------------------
extern "C" void kernel_launch(void* const* d_in, const int* in_sizes, int n_in,
                              void* d_out, int out_size, void* d_ws, size_t ws_size,
                              hipStream_t stream) {
  (void)in_sizes; (void)n_in; (void)out_size; (void)ws_size;

  const float* h   = (const float*)d_in[0];
  const float* ew  = (const float*)d_in[1];
  const int*   dst = (const int*)d_in[3];   // src (d_in[2]) implied: n*32+lane
  const float* Wq  = (const float*)d_in[4];  const float* bq  = (const float*)d_in[5];
  const float* Wk  = (const float*)d_in[6];  const float* bk  = (const float*)d_in[7];
  const float* Wv  = (const float*)d_in[8];  const float* bv  = (const float*)d_in[9];
  const float* Wo  = (const float*)d_in[10]; const float* bo  = (const float*)d_in[11];
  const float* W1  = (const float*)d_in[12]; const float* b1  = (const float*)d_in[13];
  const float* W2  = (const float*)d_in[14]; const float* b2  = (const float*)d_in[15];
  const float* g1  = (const float*)d_in[16]; const float* be1 = (const float*)d_in[17];
  const float* g2  = (const float*)d_in[18]; const float* be2 = (const float*)d_in[19];

  char* ws = (char*)d_ws;
  size_t off = 0;
  auto alloc = [&](size_t bytes) -> void* {
    void* p = ws + off;
    off = (off + bytes + 255) & ~(size_t)255;
    return p;
  };

  bf16_t* hb    = (bf16_t*)alloc((size_t)N_NODES * DMODEL * 2);
  bf16_t* wqkvb = (bf16_t*)alloc((size_t)3 * DMODEL * DMODEL * 2);
  bf16_t* wob   = (bf16_t*)alloc((size_t)DMODEL * DMODEL * 2);
  bf16_t* w1b   = (bf16_t*)alloc((size_t)DFF * DMODEL * 2);
  bf16_t* w2b   = (bf16_t*)alloc((size_t)DMODEL * DFF * 2);
  float*  bqkv  = (float*)alloc((size_t)3 * DMODEL * 4);
  bf16_t* qkvb  = (bf16_t*)alloc((size_t)N_NODES * 3 * DMODEL * 2);
  bf16_t* attnb = (bf16_t*)alloc((size_t)N_NODES * DMODEL * 2);
  float*  pre1  = (float*)alloc((size_t)N_NODES * DMODEL * 4);
  float*  h1f   = (float*)alloc((size_t)N_NODES * DMODEL * 4);
  bf16_t* h1b   = (bf16_t*)alloc((size_t)N_NODES * DMODEL * 2);
  bf16_t* zb    = (bf16_t*)alloc((size_t)N_NODES * DFF * 2);
  float*  pre2  = (float*)alloc((size_t)N_NODES * DMODEL * 4);

  auto cvt = [&](const float* s_, bf16_t* d_, int n_) {
    f32_to_bf16_kernel<<<(n_ + 255) / 256, 256, 0, stream>>>(s_, d_, n_);
  };
  cvt(h,  hb,                          N_NODES * DMODEL);
  cvt(Wq, wqkvb,                       DMODEL * DMODEL);
  cvt(Wk, wqkvb + DMODEL * DMODEL,     DMODEL * DMODEL);
  cvt(Wv, wqkvb + 2 * DMODEL * DMODEL, DMODEL * DMODEL);
  cvt(Wo, wob, DMODEL * DMODEL);
  cvt(W1, w1b, DFF * DMODEL);
  cvt(W2, w2b, DMODEL * DFF);
  pack_bias3_kernel<<<3, 256, 0, stream>>>(bq, bk, bv, bqkv);

  // qkv = h @ [Wq;Wk;Wv]^T + biases                      (bf16 out)
  gemm_bf16_wmma<8><<<dim3(768 / 128, N_NODES / 64), 256, 0, stream>>>(
      hb, wqkvb, bqkv, nullptr, nullptr, qkvb, 768, DMODEL);

  // sparse masked multi-head attention
  attn_sparse_kernel<<<(N_NODES * NHEAD) / 4, 128, 0, stream>>>(qkvb, ew, dst, attnb);

  // pre1 = attn @ Wo^T + bo + h                          (fp32 out, resid)
  gemm_bf16_wmma<2 | 4><<<dim3(DMODEL / 128, N_NODES / 64), 256, 0, stream>>>(
      attnb, wob, bo, h, pre1, nullptr, DMODEL, DMODEL);

  // h1 = LN1(pre1)  -> fp32 (residual) + bf16 (next GEMM)
  layernorm_kernel<<<N_NODES / 8, 256, 0, stream>>>(pre1, g1, be1, h1f, h1b);

  // z = relu(h1 @ W1^T + b1)                             (bf16 out)
  gemm_bf16_wmma<1 | 8><<<dim3(DFF / 128, N_NODES / 64), 256, 0, stream>>>(
      h1b, w1b, b1, nullptr, nullptr, zb, DFF, DMODEL);

  // pre2 = z @ W2^T + b2 + h1                            (fp32 out, resid)
  gemm_bf16_wmma<2 | 4><<<dim3(DMODEL / 128, N_NODES / 64), 256, 0, stream>>>(
      zb, w2b, b2, h1f, pre2, nullptr, DMODEL, DFF);

  // out = LN2(pre2)
  layernorm_kernel<<<N_NODES / 8, 256, 0, stream>>>(pre2, g2, be2, (float*)d_out, nullptr);
}
